// MultiHeadAttention_74517682586324
// MI455X (gfx1250) — matrix operand order repaired and checked by reading, not compiled, not run on hardware
//
#include <hip/hip_runtime.h>
#include <hip/hip_bf16.h>
#include <math.h>

// Problem shape (fixed by the reference)
#define B_  4
#define S_  2048
#define D_  1024
#define H_  16
#define HD_ 64
#define M_  (B_ * S_)   // 8192 rows for all projection GEMMs

typedef __attribute__((ext_vector_type(16))) _Float16 v16h;
typedef __attribute__((ext_vector_type(8)))  float    v8f;
typedef __attribute__((ext_vector_type(4)))  float    fx4;   // trivial 16B vector
typedef __attribute__((ext_vector_type(4)))  unsigned int ui4;
typedef __attribute__((ext_vector_type(8)))  unsigned int ui8;

union Frag {
    fx4  f4[2];
    v16h h;
};

static __device__ __forceinline__ v8f wmma16(const Frag& a, const Frag& b, v8f c) {
    // D = A(16x32 f16) * B(32x16 f16) + C(16x16 f32)
    return __builtin_amdgcn_wmma_f32_16x16x32_f16(false, a.h, false, b.h, (short)0, c, false, false);
}

static __device__ __forceinline__ void load_frag(Frag& f, const _Float16* p) {
    f.f4[0] = *reinterpret_cast<const fx4*>(p);
    f.f4[1] = *reinterpret_cast<const fx4*>(p + 8);
}

// Issue one TDM DMA: 64 rows x 64 cols of f16 (row stride 1024 elems) -> LDS
// (contiguous 64x64 tile).  2-SGPR-group descriptor form (<=2D tensor).
static __device__ __forceinline__ void tdm_load_b64x64(const _Float16* gptr, void* lptr) {
    const unsigned long long ga = (unsigned long long)(uintptr_t)gptr;
    const unsigned int      la  = (unsigned int)(uintptr_t)lptr;   // low 32 = LDS offset
    ui4 g0;
    g0[0] = 1u;                                           // count=1 (valid), no gather
    g0[1] = la;                                           // lds_addr
    g0[2] = (unsigned int)ga;                             // global_addr[31:0]
    g0[3] = (unsigned int)((ga >> 32) & 0x1FFFFFFu) | (2u << 30);  // addr[56:32] | type=2
    ui8 g1;
    g1[0] = 0x00010000u;          // workgroup_mask=0, data_size=1 (2 bytes)
    g1[1] = 0u;                   // tensor_dim0[15:0]=0 (dim0 = 1<<20, no clipping)
    g1[2] = 0x10u;                // tensor_dim0[31:16]=16 ; tensor_dim1[15:0]=0
    g1[3] = (64u << 16) | 0x10u;  // tensor_dim1[31:16]=16 ; tile_dim0=64 (K, contiguous)
    g1[4] = 64u;                  // tile_dim1=64 (N rows), tile_dim2=0
    g1[5] = (unsigned int)D_;     // tensor_dim0_stride = 1024 elements
    g1[6] = 0u;
    g1[7] = 0u;
    asm volatile("tensor_load_to_lds %0, %1" : : "s"(g0), "s"(g1) : "memory");
}

// ---------------------------------------------------------------- converts --
__global__ __launch_bounds__(256) void cvt_f32_to_f16(const float* __restrict__ in,
                                                      _Float16* __restrict__ out, int n) {
    int i = blockIdx.x * 256 + threadIdx.x;
    if (i < n) out[i] = (_Float16)in[i];
}

// Wt[n][k] = (f16) W[k][n]   (K = N = 1024)
__global__ __launch_bounds__(256) void cvt_transpose_w(const float* __restrict__ W,
                                                       _Float16* __restrict__ Wt) {
    int i = blockIdx.x * 256 + threadIdx.x;      // 0 .. 1024*1024-1
    int n = i >> 10;
    int k = i & 1023;
    Wt[i] = (_Float16)W[(size_t)k * D_ + n];
}

// ------------------------------------------------------------------- GEMM ---
// C[M_ x D_] = A[M_ x D_] * Bt^T  where Bt is [N][K] row-major (B transposed).
// Block = 8 waves = 128(M) x 64(N).  B tile (64x64 f16 per 64-k stage) is
// staged into LDS by the Tensor Data Mover (double buffered); A streams from
// global per wave.  mode: 0 = f16 row-major (scaled), 1 = f16 V transposed
// per head [b][h][hd][s], 2 = f32 row-major.
__global__ __launch_bounds__(256)
void gemm_f16(const _Float16* __restrict__ A,
              const _Float16* __restrict__ Bt,
              void* __restrict__ Cout,
              float scale, int mode) {
    __shared__ __align__(16) _Float16 bsm[2][64][64];   // [buf][n row][k col] = 8KB each

    const int lane = threadIdx.x & 31;
    const int wave = threadIdx.x >> 5;
    const int lh   = lane >> 4;       // lane half 0/1
    const int lm   = lane & 15;
    const int m0   = blockIdx.y * 128 + wave * 16;
    const int n0   = blockIdx.x * 64;
    const int NST  = D_ / 64;         // 16 k-stages

    v8f acc[4] = {};
    const _Float16* arow  = A  + (size_t)(m0 + lm) * D_;
    const _Float16* btile = Bt + (size_t)n0 * D_;       // row n0, col 0

    if (wave == 0) tdm_load_b64x64(btile, &bsm[0][0][0]);

    for (int st = 0; st < NST; ++st) {
        const int buf = st & 1;
        if (wave == 0) {
            if (st + 1 < NST) {
                tdm_load_b64x64(btile + (st + 1) * 64, &bsm[buf ^ 1][0][0]);
                __builtin_amdgcn_s_wait_tensorcnt(1);   // stage st's DMA complete
            } else {
                __builtin_amdgcn_s_wait_tensorcnt(0);
            }
        }
        __syncthreads();                                // buf ready for all waves

        const int kbase = st * 64;
#pragma unroll
        for (int half = 0; half < 2; ++half) {
            const int kloc = half * 32;
            Frag a;
            a.f4[0] = *reinterpret_cast<const fx4*>(arow + kbase + kloc + 16 * lh);
            a.f4[1] = *reinterpret_cast<const fx4*>(arow + kbase + kloc + 16 * lh + 8);
            __builtin_prefetch(arow + kbase + 256, 0, 1);
            // Load all 4 B fragments first, then issue 4 back-to-back WMMAs so
            // the ds loads clause together and the XDL pipe stays fed.
            Frag b0, b1, b2, b3;
            load_frag(b0, &bsm[buf][ 0 + lm][kloc + 16 * lh]);
            load_frag(b1, &bsm[buf][16 + lm][kloc + 16 * lh]);
            load_frag(b2, &bsm[buf][32 + lm][kloc + 16 * lh]);
            load_frag(b3, &bsm[buf][48 + lm][kloc + 16 * lh]);
            acc[0] = wmma16(a, b0, acc[0]);
            acc[1] = wmma16(a, b1, acc[1]);
            acc[2] = wmma16(a, b2, acc[2]);
            acc[3] = wmma16(a, b3, acc[3]);
        }
        __syncthreads();                                // reads done before overwrite
    }

    // C layout: VGPR r, lanes 0-15 -> M=r, lanes 16-31 -> M=8+r; N = lane&15.
#pragma unroll
    for (int t = 0; t < 4; ++t) {
#pragma unroll
        for (int r = 0; r < 8; ++r) {
            const int m = m0 + r + 8 * lh;
            const int n = n0 + t * 16 + lm;
            const float v = acc[t][r] * scale;
            if (mode == 2) {
                ((float*)Cout)[(size_t)m * D_ + n] = v;
            } else if (mode == 0) {
                ((_Float16*)Cout)[(size_t)m * D_ + n] = (_Float16)v;
            } else {                 // V: [b][h][hd][s]
                const int b  = m >> 11;           // m / S_
                const int s  = m & (S_ - 1);
                const int h  = n >> 6;            // n / HD_
                const int hd = n & (HD_ - 1);
                ((_Float16*)Cout)[((((size_t)b * H_ + h) * HD_) + hd) * S_ + s] = (_Float16)v;
            }
        }
    }
}

// -------------------------------------------------------- flash attention ---
// Q,K: f16 [b][s][h*64+hd] (Q pre-scaled by 1/8).  Vt: f16 [b][h][hd][s].
// Ctx out: f16 [b][s][h*64+hd].  One wave = 16 queries, online softmax over
// causal 32-key blocks.
__global__ __launch_bounds__(256)
void attn_flash(const _Float16* __restrict__ Qh,
                const _Float16* __restrict__ Kh,
                const _Float16* __restrict__ Vt,
                _Float16* __restrict__ Ctx) {
    __shared__ __align__(16) _Float16 plds[8][16][32];   // per-wave P tile 16x32

    const int lane = threadIdx.x & 31;
    const int wave = threadIdx.x >> 5;
    const int lh   = lane >> 4;
    const int lm   = lane & 15;
    const int bh   = blockIdx.y;          // b*H_ + h
    const int b    = bh >> 4;
    const int h    = bh & (H_ - 1);
    const int q0   = blockIdx.x * 128 + wave * 16;

    // Preload this wave's Q tile (16x64) in A-operand chunk layout.
    const _Float16* qbase = Qh + ((size_t)b * S_ + q0 + lm) * D_ + h * HD_;
    fx4 qreg[4];
#pragma unroll
    for (int j = 0; j < 4; ++j)
        qreg[j] = *reinterpret_cast<const fx4*>(qbase + (j >> 1) * 32 + 16 * lh + (j & 1) * 8);

    v8f o[4] = {};
    float m_i[8], l_i[8];
#pragma unroll
    for (int r = 0; r < 8; ++r) { m_i[r] = -INFINITY; l_i[r] = 0.0f; }

    const int nblk = (q0 + 16 + 31) >> 5;             // causal: keys <= q0+15
    for (int blk = 0; blk < nblk; ++blk) {
        const int k0 = blk * 32;

        // ---- scores: S = Q (16x64) x K^T (64x32) -> two 16x16 f32 tiles
        // Load all 4 K fragments, then 4 back-to-back WMMAs.
        Frag a0, a1;
        a0.f4[0] = qreg[0]; a0.f4[1] = qreg[1];
        a1.f4[0] = qreg[2]; a1.f4[1] = qreg[3];
        const _Float16* kb = Kh + ((size_t)b * S_ + k0 + lm) * D_ + h * HD_ + 16 * lh;
        Frag k00, k01, k10, k11;            // [kk][t]
        load_frag(k00, kb);
        load_frag(k01, kb + (size_t)16 * D_);
        load_frag(k10, kb + 32);
        load_frag(k11, kb + (size_t)16 * D_ + 32);
        v8f s0 = {}, s1 = {};
        s0 = wmma16(a0, k00, s0);
        s1 = wmma16(a0, k01, s1);
        s0 = wmma16(a1, k10, s0);
        s1 = wmma16(a1, k11, s1);

        // ---- causal mask (only the diagonal block can need it)
        if (blk == nblk - 1) {
#pragma unroll
            for (int r = 0; r < 8; ++r) {
                const int q = q0 + r + 8 * lh;
                if (k0 + lm      > q) s0[r] = -INFINITY;
                if (k0 + 16 + lm > q) s1[r] = -INFINITY;
            }
        }

        // ---- online softmax (row = r + 8*lh lives in a 16-lane group)
#pragma unroll
        for (int r = 0; r < 8; ++r) {
            float bm = fmaxf(s0[r], s1[r]);
#pragma unroll
            for (int off = 8; off > 0; off >>= 1)
                bm = fmaxf(bm, __shfl_xor(bm, off, 16));
            const float mn    = fmaxf(m_i[r], bm);
            const float alpha = __expf(m_i[r] - mn);
            const float p0 = __expf(s0[r] - mn);
            const float p1 = __expf(s1[r] - mn);
            float rs = p0 + p1;
#pragma unroll
            for (int off = 8; off > 0; off >>= 1)
                rs += __shfl_xor(rs, off, 16);
            l_i[r] = l_i[r] * alpha + rs;
            m_i[r] = mn;
            o[0][r] = o[0][r] * alpha;
            o[1][r] = o[1][r] * alpha;
            o[2][r] = o[2][r] * alpha;
            o[3][r] = o[3][r] * alpha;
            s0[r] = p0;
            s1[r] = p1;
        }

        // ---- repack P (C-layout f32) -> A-layout f16 via LDS
#pragma unroll
        for (int r = 0; r < 8; ++r) {
            const int row = r + 8 * lh;
            plds[wave][row][lm]      = (_Float16)s0[r];
            plds[wave][row][16 + lm] = (_Float16)s1[r];
        }
        __builtin_amdgcn_wave_barrier();
        Frag pa;
        pa.f4[0] = *reinterpret_cast<const fx4*>(&plds[wave][lm][16 * lh]);
        pa.f4[1] = *reinterpret_cast<const fx4*>(&plds[wave][lm][16 * lh + 8]);
        __builtin_amdgcn_wave_barrier();

        // ---- ctx accumulate: O += P (16x32) x V (32x64)
        const _Float16* vb = Vt + ((size_t)bh * HD_ + lm) * S_ + k0 + 16 * lh;
        Frag v0, v1, v2, v3;
        load_frag(v0, vb);
        load_frag(v1, vb + (size_t)16 * S_);
        load_frag(v2, vb + (size_t)32 * S_);
        load_frag(v3, vb + (size_t)48 * S_);
        o[0] = wmma16(pa, v0, o[0]);
        o[1] = wmma16(pa, v1, o[1]);
        o[2] = wmma16(pa, v2, o[2]);
        o[3] = wmma16(pa, v3, o[3]);
    }

    // ---- normalize and store ctx (f16, [b][s][h*64+hd])
#pragma unroll
    for (int r = 0; r < 8; ++r) {
        const float inv = 1.0f / l_i[r];
        const int   q   = q0 + r + 8 * lh;
        _Float16* crow = Ctx + ((size_t)b * S_ + q) * D_ + h * HD_ + lm;
        crow[0]  = (_Float16)(o[0][r] * inv);
        crow[16] = (_Float16)(o[1][r] * inv);
        crow[32] = (_Float16)(o[2][r] * inv);
        crow[48] = (_Float16)(o[3][r] * inv);
    }
}

// ------------------------------------------------------------------ launch --
extern "C" void kernel_launch(void* const* d_in, const int* in_sizes, int n_in,
                              void* d_out, int out_size, void* d_ws, size_t ws_size,
                              hipStream_t stream) {
    const float* x  = (const float*)d_in[0];
    const float* Wq = (const float*)d_in[1];
    const float* Wk = (const float*)d_in[2];
    const float* Wv = (const float*)d_in[3];
    const float* Wo = (const float*)d_in[4];

    uint8_t* ws = (uint8_t*)d_ws;
    const size_t SZ_X  = (size_t)M_ * D_ * sizeof(_Float16);   // 16 MB
    const size_t SZ_W  = (size_t)D_ * D_ * sizeof(_Float16);   //  2 MB
    _Float16* Xh   = (_Float16*)(ws);
    _Float16* Qh   = (_Float16*)(ws + SZ_X);
    _Float16* Kh   = (_Float16*)(ws + 2 * SZ_X);
    _Float16* Vth  = (_Float16*)(ws + 3 * SZ_X);
    _Float16* Ctxh = (_Float16*)(ws + 4 * SZ_X);
    _Float16* Wqh  = (_Float16*)(ws + 5 * SZ_X);
    _Float16* Wkh  = (_Float16*)(ws + 5 * SZ_X + SZ_W);
    _Float16* Wvh  = (_Float16*)(ws + 5 * SZ_X + 2 * SZ_W);
    _Float16* Woh  = (_Float16*)(ws + 5 * SZ_X + 3 * SZ_W);

    // 1) convert inputs to f16 (weights transposed so WMMA B-operand is contiguous)
    cvt_f32_to_f16<<<(M_ * D_ + 255) / 256, 256, 0, stream>>>(x, Xh, M_ * D_);
    cvt_transpose_w<<<(D_ * D_) / 256, 256, 0, stream>>>(Wq, Wqh);
    cvt_transpose_w<<<(D_ * D_) / 256, 256, 0, stream>>>(Wk, Wkh);
    cvt_transpose_w<<<(D_ * D_) / 256, 256, 0, stream>>>(Wv, Wvh);
    cvt_transpose_w<<<(D_ * D_) / 256, 256, 0, stream>>>(Wo, Woh);

    // 2) projections: Q (scaled by 1/sqrt(64)), K, V (stored [b][h][hd][s])
    dim3 ggrid(D_ / 64, M_ / 128);
    gemm_f16<<<ggrid, 256, 0, stream>>>(Xh, Wqh, Qh, 0.125f, 0);
    gemm_f16<<<ggrid, 256, 0, stream>>>(Xh, Wkh, Kh, 1.0f, 0);
    gemm_f16<<<ggrid, 256, 0, stream>>>(Xh, Wvh, Vth, 1.0f, 1);

    // 3) causal flash attention
    dim3 agrid(S_ / 128, B_ * H_);
    attn_flash<<<agrid, 256, 0, stream>>>(Qh, Kh, Vth, Ctxh);

    // 4) output projection -> f32 d_out
    gemm_f16<<<ggrid, 256, 0, stream>>>(Ctxh, Woh, d_out, 1.0f, 2);
}
